// AttentionRNN_18580028523223
// MI455X (gfx1250) — compile-verified, matrix operand order checked by reading, not run on hardware
//
#include <hip/hip_runtime.h>
#include <hip/hip_bf16.h>

// ---------------------------------------------------------------------------
// AttentionRNN (bi-LSTM + additive attention + linear) for gfx1250 / MI455X.
// All large GEMMs use v_wmma_f32_16x16x32_bf16 with fragment-prepacked bf16
// operands (32 contiguous bytes per lane per fragment). xg is stored in a
// D-fragment (lane-contiguous) layout so the serial LSTM loop loads its
// accumulator init with b128 vector loads.
// ---------------------------------------------------------------------------

#define T_SEQ 512
#define BATCH 64
#define EMB   256
#define HID   256
#define G4H   1024      // 4*HID
#define D2    512       // 2*HID
#define NCLS  5
#define MROWS (T_SEQ * BATCH)   // 32768

typedef __attribute__((ext_vector_type(16))) __bf16          v16bf;
typedef __attribute__((ext_vector_type(8)))  float           v8f;
typedef __attribute__((ext_vector_type(8)))  unsigned short  ushort8;

__device__ __forceinline__ unsigned short f2bf(float f) {
    unsigned u = __float_as_uint(f);
    return (unsigned short)((u + 0x7FFFu + ((u >> 16) & 1u)) >> 16);  // RNE
}
__device__ __forceinline__ float bf2f(unsigned short h) {
    return __uint_as_float(((unsigned)h) << 16);
}
__device__ __forceinline__ unsigned pack2bf(float lo, float hi) {
    return (unsigned)f2bf(lo) | ((unsigned)f2bf(hi) << 16);
}
// 16-bit WMMA fragment k-map within a 32-wide k-tile (ISA 7.12.2):
// lanes 0-15 hold k [0,8) and [16,24); lanes 16-31 hold k [8,16) and [24,32),
// two consecutive k per dword.
__device__ __forceinline__ int kbase(int half, int v) {
    return (v < 4 ? 0 : 16) + (half ? 8 : 0) + (v & 3) * 2;
}
__device__ __forceinline__ void kmap(int ko, int& half, int& v) {
    int grp = ko >> 3;                 // 0..3
    half = grp & 1;
    v = ((grp >> 1) << 2) + ((ko & 7) >> 1);
}
__device__ __forceinline__ float sigm(float x) {
    return 1.f / (1.f + __expf(-x));
}
__device__ __forceinline__ float tanh_fast(float x) {
    x = fminf(fmaxf(x, -15.f), 15.f);
    float e = __expf(2.f * x);
    return (e - 1.f) / (e + 1.f);
}

// ---------------------------------------------------------------------------
// Pack a row-major fp32 matrix into bf16 WMMA fragments.
// Logical B[k][n] = transpose ? src[n*K+k] : src[k*N+n].
// Fragment tile (nt,kt) stored nt-major: tileIdx = nt*(K/32)+kt,
// element (lane, dword v, pair p) at dst[tileIdx*512 + lane*16 + v*2 + p].
// One thread per dword pair (single b32 store).
__global__ void k_pack_frag(unsigned short* __restrict__ dst,
                            const float* __restrict__ src,
                            int K, int N, int transpose) {
    long tid = (long)blockIdx.x * blockDim.x + threadIdx.x;
    int KT = K >> 5;
    long total = (long)KT * (N >> 4) * 256;
    if (tid >= total) return;
    int v    = tid & 7;
    int lane = (tid >> 3) & 31;
    long tile = tid >> 8;
    int kt = (int)(tile % KT);
    long nt = tile / KT;
    int half = lane >> 4;
    long n = nt * 16 + (lane & 15);
    int k0 = kt * 32 + kbase(half, v);
    float f0 = transpose ? src[n * K + k0]     : src[(long)k0 * N + n];
    float f1 = transpose ? src[n * K + k0 + 1] : src[(long)(k0 + 1) * N + n];
    *(unsigned*)(dst + tile * 512 + lane * 16 + v * 2) = pack2bf(f0, f1);
}

// Embedding gather fused with A-fragment packing: x (M=T*B, K=256) bf16.
__global__ void k_gather_pack_x(unsigned short* __restrict__ dst,
                                const int* __restrict__ embed,
                                const float* __restrict__ table) {
    long tid = (long)blockIdx.x * blockDim.x + threadIdx.x;   // M*K/2 dwords
    int v    = tid & 7;
    int lane = (tid >> 3) & 31;
    long tile = tid >> 8;
    int kt   = (int)(tile & 7);          // K/32 = 8
    long mt  = tile >> 3;
    int half = lane >> 4;
    long m = mt * 16 + (lane & 15);
    int k0 = kt * 32 + kbase(half, v);
    const float* r = table + (long)embed[m] * EMB;
    *(unsigned*)(dst + tile * 512 + lane * 16 + v * 2) = pack2bf(r[k0], r[k0 + 1]);
}

// ---------------------------------------------------------------------------
// xg[dir] = x @ Wih[dir]^T + bih + bhh
// Stored in D-fragment layout: tile (mtile, nt) at (mtile*64 + nt)*256,
// lane's 8 values contiguous at + lane*8  -> one b128 store / load per tile.
// grid (M/64, 16, 2), block 128 (4 waves, one 16x64 strip each)
__global__ void k_gemm_xg(unsigned short* __restrict__ xg,
                          const unsigned short* __restrict__ xpk,
                          const unsigned short* __restrict__ wpk_f,
                          const unsigned short* __restrict__ wpk_b,
                          const float* __restrict__ bih_f, const float* __restrict__ bhh_f,
                          const float* __restrict__ bih_b, const float* __restrict__ bhh_b) {
    int dir = blockIdx.z;
    const unsigned short* wpk = dir ? wpk_b : wpk_f;
    const float* bi = dir ? bih_b : bih_f;
    const float* bh = dir ? bhh_b : bhh_f;
    unsigned short* out = xg + (size_t)dir * MROWS * G4H;
    int lane = threadIdx.x & 31, w = threadIdx.x >> 5;
    long mtile = (long)blockIdx.x * 4 + w;
    int nb = blockIdx.y;
    int ln = lane & 15;

    v8f acc[4];
    #pragma unroll
    for (int g = 0; g < 4; ++g) {
        int n = (nb * 4 + g) * 16 + ln;
        float b = bi[n] + bh[n];
        #pragma unroll
        for (int r = 0; r < 8; ++r) acc[g][r] = b;
    }
    #pragma unroll
    for (int kt = 0; kt < 8; ++kt) {
        v16bf a = *(const v16bf*)(xpk + (mtile * 8 + kt) * 512 + lane * 16);
        #pragma unroll
        for (int g = 0; g < 4; ++g) {
            long nt = nb * 4 + g;
            const unsigned short* bp = wpk + (nt * 8 + kt) * 512 + lane * 16;
            if (kt < 7) __builtin_prefetch(bp + 512, 0, 1);
            v16bf b = *(const v16bf*)bp;
            acc[g] = __builtin_amdgcn_wmma_f32_16x16x32_bf16(
                false, a, false, b, (short)0, acc[g], false, false);
        }
    }
    #pragma unroll
    for (int g = 0; g < 4; ++g) {
        ushort8 hv;
        #pragma unroll
        for (int r = 0; r < 8; ++r) hv[r] = f2bf(acc[g][r]);
        *(ushort8*)(out + (mtile * 64 + nb * 4 + g) * 256 + lane * 8) = hv;
    }
}

// ---------------------------------------------------------------------------
// LSTM recurrence, one persistent workgroup (32 waves) per direction.
// h lives in LDS as packed bf16 A-fragments; c stays in registers.
// Each wave owns batch-tile mt (0..3) and hidden-tiles {jt0, jt0+8}; for each
// it accumulates all 4 gate tiles so i/f/g/o combine elementwise in registers.
// Output h written directly as packed bf16 A-fragments of out (M, K=512).
__global__ void k_lstm(unsigned short* __restrict__ outpk,
                       const unsigned short* __restrict__ xg,
                       const unsigned short* __restrict__ whh_f,
                       const unsigned short* __restrict__ whh_b,
                       const float* __restrict__ h0, const float* __restrict__ c0) {
    int dir = blockIdx.x;
    const unsigned short* wpk = dir ? whh_b : whh_f;
    const unsigned short* xgd = xg + (size_t)dir * MROWS * G4H;
    __shared__ __align__(32) unsigned short hbuf[4 * 8 * 32 * 16];   // 32 KB

    int lane = threadIdx.x & 31, w = threadIdx.x >> 5;
    int mt  = w >> 3;            // 0..3
    int jt0 = w & 7;             // tasks jt0, jt0+8
    int half = lane >> 4, ln = lane & 15;

    // init h fragments from h0[dir]
    for (int i = threadIdx.x; i < 4 * 8 * 32 * 8; i += 1024) {
        int v = i & 7, lane_ = (i >> 3) & 31, tile = i >> 8;
        int kt = tile & 7, mtl = tile >> 3;
        int hf = lane_ >> 4;
        int b = mtl * 16 + (lane_ & 15);
        int k0 = kt * 32 + kbase(hf, v);
        const float* hr = h0 + ((long)dir * BATCH + b) * HID;
        hbuf[tile * 512 + lane_ * 16 + v * 2]     = f2bf(hr[k0]);
        hbuf[tile * 512 + lane_ * 16 + v * 2 + 1] = f2bf(hr[k0 + 1]);
    }
    float cst[2][8];
    #pragma unroll
    for (int s = 0; s < 2; ++s) {
        int jt = jt0 + s * 8;
        #pragma unroll
        for (int r = 0; r < 8; ++r) {
            int b = mt * 16 + r + half * 8;
            cst[s][r] = c0[((long)dir * BATCH + b) * HID + jt * 16 + ln];
        }
    }
    __syncthreads();

    for (int t = 0; t < T_SEQ; ++t) {
        int tt = dir ? (T_SEQ - 1 - t) : t;
        v8f acc[2][4];
        // C-init from precomputed xg: one b128 load per gate-tile per lane
        #pragma unroll
        for (int s = 0; s < 2; ++s) {
            int jt = jt0 + s * 8;
            #pragma unroll
            for (int g = 0; g < 4; ++g) {
                long nt = g * 16 + jt;
                ushort8 xv = *(const ushort8*)(
                    xgd + (((long)tt * 4 + mt) * 64 + nt) * 256 + lane * 8);
                #pragma unroll
                for (int r = 0; r < 8; ++r) acc[s][g][r] = bf2f(xv[r]);
            }
        }
        // h @ Whh^T : K = 256 -> 8 k-tiles, 64 WMMAs per wave per step
        #pragma unroll
        for (int kt = 0; kt < 8; ++kt) {
            v16bf a = *(const v16bf*)(hbuf + (mt * 8 + kt) * 512 + lane * 16);
            #pragma unroll
            for (int s = 0; s < 2; ++s) {
                int jt = jt0 + s * 8;
                #pragma unroll
                for (int g = 0; g < 4; ++g) {
                    long nt = g * 16 + jt;
                    const unsigned short* bp = wpk + (nt * 8 + kt) * 512 + lane * 16;
                    __builtin_prefetch(bp + 512, 0, 1);
                    v16bf b = *(const v16bf*)bp;
                    acc[s][g] = __builtin_amdgcn_wmma_f32_16x16x32_bf16(
                        false, a, false, b, (short)0, acc[s][g], false, false);
                }
            }
        }
        __syncthreads();   // all reads of h(t-1) complete before overwrite

        #pragma unroll
        for (int s = 0; s < 2; ++s) {
            int jt = jt0 + s * 8;
            int ktile = jt >> 1;
            int ko = (jt & 1) * 16 + ln;
            int hf2, v2; kmap(ko, hf2, v2);
            #pragma unroll
            for (int r = 0; r < 8; ++r) {
                float ig = sigm(acc[s][0][r]);
                float fg = sigm(acc[s][1][r]);
                float gg = tanh_fast(acc[s][2][r]);
                float og = sigm(acc[s][3][r]);
                float cn = fg * cst[s][r] + ig * gg;
                cst[s][r] = cn;
                unsigned short hb = f2bf(og * tanh_fast(cn));
                int bloc = mt * 16 + r + half * 8;            // local batch row
                int lane2 = hf2 * 16 + (bloc & 15);
                // LDS fragment for next step
                hbuf[(mt * 8 + ktile) * 512 + lane2 * 16 + v2 * 2 + (ko & 1)] = hb;
                // global packed A-fragment of out (k offset dir*256)
                long Mtg = (long)tt * 4 + mt;
                outpk[((Mtg * 16 + dir * 8 + ktile) * 32 + lane2) * 16 + v2 * 2 + (ko & 1)] = hb;
            }
        }
        __syncthreads();   // h(t) visible to all waves
    }
}

// ---------------------------------------------------------------------------
// scores[m] += sum_n tanh((out @ W_word)[m,n] + b_word[n]) * proj[n]
// grid (M/64, 8), block 128; scores pre-zeroed.
__global__ void k_attn_score(float* __restrict__ scores,
                             const unsigned short* __restrict__ outpk,
                             const unsigned short* __restrict__ wwpk,
                             const float* __restrict__ b_word,
                             const float* __restrict__ proj) {
    int lane = threadIdx.x & 31, w = threadIdx.x >> 5;
    long mtile = (long)blockIdx.x * 4 + w;
    int nb = blockIdx.y;
    int half = lane >> 4, ln = lane & 15;

    v8f acc[4];
    #pragma unroll
    for (int g = 0; g < 4; ++g)
        #pragma unroll
        for (int r = 0; r < 8; ++r) acc[g][r] = 0.f;

    #pragma unroll
    for (int kt = 0; kt < 16; ++kt) {
        v16bf a = *(const v16bf*)(outpk + (mtile * 16 + kt) * 512 + lane * 16);
        #pragma unroll
        for (int g = 0; g < 4; ++g) {
            long nt = nb * 4 + g;
            const unsigned short* bp = wwpk + (nt * 16 + kt) * 512 + lane * 16;
            if (kt < 15) __builtin_prefetch(bp + 512, 0, 1);
            v16bf b = *(const v16bf*)bp;
            acc[g] = __builtin_amdgcn_wmma_f32_16x16x32_bf16(
                false, a, false, b, (short)0, acc[g], false, false);
        }
    }
    float part[8];
    #pragma unroll
    for (int r = 0; r < 8; ++r) part[r] = 0.f;
    #pragma unroll
    for (int g = 0; g < 4; ++g) {
        int n = (nb * 4 + g) * 16 + ln;
        float bw = b_word[n], pj = proj[n];
        #pragma unroll
        for (int r = 0; r < 8; ++r)
            part[r] += tanh_fast(acc[g][r] + bw) * pj;
    }
    // reduce over the 16 lanes of each half (same m, varying n)
    #pragma unroll
    for (int off = 1; off < 16; off <<= 1)
        #pragma unroll
        for (int r = 0; r < 8; ++r)
            part[r] += __shfl_xor(part[r], off);
    if (ln == 0) {
        #pragma unroll
        for (int r = 0; r < 8; ++r)
            atomicAdd(&scores[mtile * 16 + r + half * 8], part[r]);
    }
}

__global__ void k_zero(float* p, long n) {
    long i = (long)blockIdx.x * blockDim.x + threadIdx.x;
    if (i < n) p[i] = 0.f;
}

// softmax over T per batch column: scores/attn are (T,B) flattened
__global__ void k_softmax(float* __restrict__ attn, const float* __restrict__ scores) {
    int b = threadIdx.x;
    if (b >= BATCH) return;
    float mx = -1e30f;
    for (int t = 0; t < T_SEQ; ++t) mx = fmaxf(mx, scores[t * BATCH + b]);
    float sum = 0.f;
    for (int t = 0; t < T_SEQ; ++t) {
        float e = __expf(scores[t * BATCH + b] - mx);
        attn[t * BATCH + b] = e;
        sum += e;
    }
    float inv = 1.f / sum;
    for (int t = 0; t < T_SEQ; ++t) attn[t * BATCH + b] *= inv;
}

// vec[b,d] = sum_t attn[t,b] * out[t,b,d]  (out read from packed bf16 frags)
__global__ void k_vec(float* __restrict__ vec, const float* __restrict__ attn,
                      const unsigned short* __restrict__ outpk) {
    int idx = blockIdx.x * blockDim.x + threadIdx.x;   // b*512 + d
    int b = idx >> 9, d = idx & 511;
    int ktile = d >> 5, ko = d & 31;
    int hf, v; kmap(ko, hf, v);
    float s = 0.f;
    for (int t = 0; t < T_SEQ; ++t) {
        long m = (long)t * BATCH + b;
        long Mt = m >> 4;
        int lane2 = hf * 16 + (int)(m & 15);
        float hv = bf2f(outpk[((Mt * 16 + ktile) * 32 + lane2) * 16 + v * 2 + (ko & 1)]);
        s += attn[t * BATCH + b] * hv;
    }
    vec[idx] = s;
}

__global__ void k_logits(float* __restrict__ out, const float* __restrict__ vec,
                         const float* __restrict__ linW, const float* __restrict__ linb) {
    int idx = blockIdx.x * blockDim.x + threadIdx.x;
    if (idx >= BATCH * NCLS) return;
    int b = idx / NCLS, c = idx % NCLS;
    float s = linb[c];
    for (int e = 0; e < D2; ++e) s += vec[b * D2 + e] * linW[c * D2 + e];
    out[idx] = s;
}

// ---------------------------------------------------------------------------
// Workspace layout (bytes), all 4 KB aligned; total ~188 MB.
static const size_t OFF_XPK   = 0;                          // 16 MB  packed x
static const size_t OFF_WIHF  = OFF_XPK   + (size_t)MROWS * EMB * 2;
static const size_t OFF_WIHB  = OFF_WIHF  + 524288;
static const size_t OFF_WHHF  = OFF_WIHB  + 524288;
static const size_t OFF_WHHB  = OFF_WHHF  + 524288;
static const size_t OFF_WW    = OFF_WHHB  + 524288;
static const size_t OFF_XG    = OFF_WW    + 524288;         // 128 MB (2 dirs, bf16)
static const size_t OFF_OUTPK = OFF_XG    + (size_t)2 * MROWS * G4H * 2;  // 32 MB
static const size_t OFF_SC    = OFF_OUTPK + (size_t)(MROWS / 16) * 16 * 512 * 2;
static const size_t OFF_AT    = OFF_SC    + MROWS * 4;
static const size_t OFF_VEC   = OFF_AT    + MROWS * 4;

extern "C" void kernel_launch(void* const* d_in, const int* in_sizes, int n_in,
                              void* d_out, int out_size, void* d_ws, size_t ws_size,
                              hipStream_t stream) {
    const int*   embed   = (const int*)  d_in[0];
    const float* h0      = (const float*)d_in[1];
    const float* c0      = (const float*)d_in[2];
    const float* emb_tab = (const float*)d_in[3];
    const float* Wih_f   = (const float*)d_in[4];
    const float* Whh_f   = (const float*)d_in[5];
    const float* bih_f   = (const float*)d_in[6];
    const float* bhh_f   = (const float*)d_in[7];
    const float* Wih_b   = (const float*)d_in[8];
    const float* Whh_b   = (const float*)d_in[9];
    const float* bih_b   = (const float*)d_in[10];
    const float* bhh_b   = (const float*)d_in[11];
    const float* W_word  = (const float*)d_in[12];
    const float* b_word  = (const float*)d_in[13];
    const float* proj    = (const float*)d_in[14];
    const float* lin_W   = (const float*)d_in[15];
    const float* lin_b   = (const float*)d_in[16];
    float* out = (float*)d_out;

    char* ws = (char*)d_ws;
    unsigned short* xpk   = (unsigned short*)(ws + OFF_XPK);
    unsigned short* wihf  = (unsigned short*)(ws + OFF_WIHF);
    unsigned short* wihb  = (unsigned short*)(ws + OFF_WIHB);
    unsigned short* whhf  = (unsigned short*)(ws + OFF_WHHF);
    unsigned short* whhb  = (unsigned short*)(ws + OFF_WHHB);
    unsigned short* wwpk  = (unsigned short*)(ws + OFF_WW);
    unsigned short* xg    = (unsigned short*)(ws + OFF_XG);
    unsigned short* outpk = (unsigned short*)(ws + OFF_OUTPK);
    float* scores = (float*)(ws + OFF_SC);
    float* attn   = (float*)(ws + OFF_AT);
    float* vec    = (float*)(ws + OFF_VEC);

    // 1) gather + pack x: M*K/2 dwords
    k_gather_pack_x<<<(MROWS * EMB / 2 + 255) / 256, 256, 0, stream>>>(xpk, embed, emb_tab);
    // 2) pack weights into B-fragments (Wih/Whh transposed, W_word direct)
    k_pack_frag<<<512, 256, 0, stream>>>(wihf, Wih_f, EMB, G4H, 1);
    k_pack_frag<<<512, 256, 0, stream>>>(wihb, Wih_b, EMB, G4H, 1);
    k_pack_frag<<<512, 256, 0, stream>>>(whhf, Whh_f, HID, G4H, 1);
    k_pack_frag<<<512, 256, 0, stream>>>(whhb, Whh_b, HID, G4H, 1);
    k_pack_frag<<<512, 256, 0, stream>>>(wwpk, W_word, D2, D2, 0);
    // 3) xg = x @ Wih^T + biases (both directions), D-fragment layout
    k_gemm_xg<<<dim3(MROWS / 64, 16, 2), 128, 0, stream>>>(
        xg, xpk, wihf, wihb, bih_f, bhh_f, bih_b, bhh_b);
    // 4) recurrence (one WGP-resident workgroup per direction)
    k_lstm<<<2, 1024, 0, stream>>>(outpk, xg, whhf, whhb, h0, c0);
    // 5) attention scores (fused tanh + proj reduction)
    k_zero<<<(MROWS + 255) / 256, 256, 0, stream>>>(scores, MROWS);
    k_attn_score<<<dim3(MROWS / 64, 8), 128, 0, stream>>>(scores, outpk, wwpk, b_word, proj);
    // 6) softmax over T, weighted sum, classifier
    k_softmax<<<1, 64, 0, stream>>>(attn, scores);
    k_vec<<<BATCH * D2 / 256, 256, 0, stream>>>(vec, attn, outpk);
    k_logits<<<2, 256, 0, stream>>>(out, vec, lin_W, lin_b);
}
